// GraphAttentionNetwork_55233279426988
// MI455X (gfx1250) — compile-verified, matrix operand order checked by reading, not run on hardware
//
#include <hip/hip_runtime.h>
#include <hip/hip_bf16.h>

// ---------------------------------------------------------------------------
// GraphAttentionNetwork on gfx1250 (MI455X): f16 WMMA GEMMs + atomic scatters
// Fragments are loaded as two contiguous 16-byte chunks (ISA 7.12.2 layout).
// ---------------------------------------------------------------------------

typedef _Float16 v16h __attribute__((ext_vector_type(16)));
typedef _Float16 v8h  __attribute__((ext_vector_type(8)));
typedef float    v8f  __attribute__((ext_vector_type(8)));

#define B_   2
#define E_   160000
#define N_   10000
#define D_   128

// ----------------------------- device helpers ------------------------------

__device__ __forceinline__ v8f wmma_f16(v16h a, v16h b, v8f c) {
  return __builtin_amdgcn_wmma_f32_16x16x32_f16(false, a, false, b, (short)0, c,
                                                false, false);
}

// Fragment load: the 16-bit A/B K-pattern (ISA 7.12.2) is two contiguous
// 8-half (16B) runs at [kbase + half*8] and [kbase + 16 + half*8].
// Works for A from row-major f16 (stride = padded K or N) and for B from a
// transposed row-major f16 weight WT[n][Kpad].
__device__ __forceinline__ v16h frag16(const _Float16* p /*row base*/,
                                       int kbase, int half) {
  const v8h* q = (const v8h*)(p + kbase + half * 8);
  v8h lo = q[0];           // k = kbase + half*8 .. +8
  v8h hi = *(const v8h*)(p + kbase + 16 + half * 8);
  v16h r;
#pragma unroll
  for (int j = 0; j < 8; ++j) { r[j] = lo[j]; r[j + 8] = hi[j]; }
  return r;
}

__device__ __forceinline__ float hsum16(float x) {
  x += __shfl_xor(x, 1, 32);
  x += __shfl_xor(x, 2, 32);
  x += __shfl_xor(x, 4, 32);
  x += __shfl_xor(x, 8, 32);
  return x;
}

__device__ __forceinline__ float gelu_f(float x) {
  return 0.5f * x * (1.0f + erff(x * 0.70710678118654752f));
}

// Monotonic float <-> uint mapping (for atomic segment-max).
__device__ __forceinline__ unsigned f2mono(float f) {
  unsigned b = __float_as_uint(f);
  return (b & 0x80000000u) ? ~b : (b | 0x80000000u);
}
__device__ __forceinline__ float mono2f(unsigned u) {
  return (u & 0x80000000u) ? __uint_as_float(u & 0x7fffffffu)
                           : __uint_as_float(~u);
}

// Row-wise LayerNorm over D=128 for a wave's 16x128 C-layout tile (8 v8f).
__device__ __forceinline__ void ln_rows(v8f* c, const float* __restrict__ g,
                                        const float* __restrict__ be, int lane) {
  int col = lane & 15;
  float s[8], q[8];
#pragma unroll
  for (int e = 0; e < 8; ++e) { s[e] = 0.f; q[e] = 0.f; }
#pragma unroll
  for (int t = 0; t < 8; ++t)
#pragma unroll
    for (int e = 0; e < 8; ++e) { float v = c[t][e]; s[e] += v; q[e] += v * v; }
#pragma unroll
  for (int e = 0; e < 8; ++e) { s[e] = hsum16(s[e]); q[e] = hsum16(q[e]); }
  float mu[8], rs[8];
#pragma unroll
  for (int e = 0; e < 8; ++e) {
    mu[e] = s[e] * (1.f / 128.f);
    float var = q[e] * (1.f / 128.f) - mu[e] * mu[e];
    rs[e] = rsqrtf(var + 1e-5f);
  }
#pragma unroll
  for (int t = 0; t < 8; ++t) {
    float gg = g[t * 16 + col], bb = be[t * 16 + col];
#pragma unroll
    for (int e = 0; e < 8; ++e)
      c[t][e] = (c[t][e] - mu[e]) * rs[e] * gg + bb;
  }
}

// ------------------------------ prep kernels -------------------------------

// Repack (K x N) f32 weight -> transposed row-major f16 WT[n][Kpad],
// zero-padding K up to Kpad (multiple of 32) so fragment loads need no guards.
__global__ void k_repack_wT(const float* __restrict__ w,
                            _Float16* __restrict__ wt, int K, int N, int Kpad) {
  int i = blockIdx.x * blockDim.x + threadIdx.x;
  if (i >= N * Kpad) return;
  int n = i / Kpad, k = i % Kpad;
  wt[i] = (k < K) ? (_Float16)w[(long)k * N + n] : (_Float16)0.f;
}

__global__ void k_fill_u32(unsigned* __restrict__ p, unsigned v, int n) {
  int i = blockIdx.x * blockDim.x + threadIdx.x;
  if (i < n) p[i] = v;
}

__global__ void k_deg(const int* __restrict__ src, const int* __restrict__ dst,
                      float* __restrict__ deg) {
  int i = blockIdx.x * blockDim.x + threadIdx.x;
  if (i >= 2 * E_) return;
  int node = (i < E_) ? src[i] : dst[i - E_];
  atomicAdd(&deg[node], 1.0f);
}

// -------------------------------- encoder ----------------------------------
// Block = 128 threads = 4 waves; wave handles a 16-edge tile (40 WMMAs).

__global__ __launch_bounds__(128) void k_encoder(
    const float* __restrict__ ef, const _Float16* __restrict__ w1t,
    const float* __restrict__ b1, const float* __restrict__ g1,
    const float* __restrict__ be1, const _Float16* __restrict__ w2t,
    const float* __restrict__ b2, const float* __restrict__ g2,
    const float* __restrict__ be2, float* __restrict__ edge_emb) {
  __shared__ __attribute__((aligned(16))) _Float16 lds_a[4][16][32];
  __shared__ __attribute__((aligned(16))) _Float16 lds_h[4][16][D_];

  int wave = threadIdx.x >> 5;
  int lane = threadIdx.x & 31;
  int half = lane >> 4;
  int m = lane & 15;
  long row0 = ((long)blockIdx.x * 4 + wave) * 16;

  // Stage 16x24 f32 rows as padded 16x32 f16 in LDS (coalesced loads).
  {
    int k = lane;                 // constant per thread over iterations
    bool valid = (k < 24);
#pragma unroll
    for (int r = 0; r < 16; ++r)
      lds_a[wave][r][k] =
          valid ? (_Float16)ef[(row0 + r) * 24 + k] : (_Float16)0.f;
  }
  // Speculative prefetch of the next tile's feature rows (dropped if OOB).
  __builtin_prefetch(ef + (row0 + 16) * 24, 0, 1);
  __syncthreads();

  // ---- GEMM1: (16x32pad) @ (32x128) ----
  v16h a = frag16(&lds_a[wave][m][0], 0, half);
  v8f c[8];
#pragma unroll
  for (int t = 0; t < 8; ++t) {
    v8f z = {};
    v16h bf = frag16(w1t + (long)(t * 16 + m) * 32, 0, half);
    c[t] = wmma_f16(a, bf, z);
    float bb = b1[t * 16 + m];
#pragma unroll
    for (int e = 0; e < 8; ++e) c[t][e] += bb;
  }
  ln_rows(c, g1, be1, lane);

  // GELU and restage as f16 A-operand for GEMM2.
#pragma unroll
  for (int t = 0; t < 8; ++t)
#pragma unroll
    for (int e = 0; e < 8; ++e)
      lds_h[wave][half * 8 + e][t * 16 + m] = (_Float16)gelu_f(c[t][e]);
  __syncthreads();

  // ---- GEMM2: (16x128) @ (128x128) ----
  v16h a2[4];
#pragma unroll
  for (int q = 0; q < 4; ++q)
    a2[q] = frag16(&lds_h[wave][m][0], 32 * q, half);

  v8f d[8];
#pragma unroll
  for (int t = 0; t < 8; ++t) {
    v8f acc = {};
#pragma unroll
    for (int q = 0; q < 4; ++q) {
      v16h bf = frag16(w2t + (long)(t * 16 + m) * 128, 32 * q, half);
      acc = wmma_f16(a2[q], bf, acc);
    }
    float bb = b2[t * 16 + m];
#pragma unroll
    for (int e = 0; e < 8; ++e) acc[e] += bb;
    d[t] = acc;
  }
  ln_rows(d, g2, be2, lane);

#pragma unroll
  for (int t = 0; t < 8; ++t)
#pragma unroll
    for (int e = 0; e < 8; ++e)
      edge_emb[(row0 + half * 8 + e) * D_ + t * 16 + m] = d[t][e];
}

// ------------------------------- attention ---------------------------------

__global__ void k_att_score(const float* __restrict__ node_emb,
                            const float* __restrict__ edge_emb,
                            const int* __restrict__ src,
                            const int* __restrict__ dst,
                            const float* __restrict__ watt,
                            const float* __restrict__ batt,
                            float* __restrict__ sbuf,
                            unsigned* __restrict__ mmax) {
  long i = (long)blockIdx.x * blockDim.x + threadIdx.x;
  const long P = (long)B_ * 2 * E_;
  if (i >= P) return;
  int b = (int)(i / (2 * E_));
  int p = (int)(i % (2 * E_));
  int node = (p < E_) ? src[p] : dst[p - E_];
  int e = (p < E_) ? p : p - E_;

  const float4* h = (const float4*)(node_emb + ((long)b * N_ + node) * D_);
  const float4* ee = (const float4*)(edge_emb + ((long)b * E_ + e) * D_);
  const float4* w0 = (const float4*)watt;
  const float4* w1 = (const float4*)(watt + D_);
  float s = batt[0];
#pragma unroll 8
  for (int k = 0; k < D_ / 4; ++k) {
    float4 av = h[k], wa = w0[k];
    float4 cv = ee[k], wb = w1[k];
    s += av.x * wa.x + av.y * wa.y + av.z * wa.z + av.w * wa.w;
    s += cv.x * wb.x + cv.y * wb.y + cv.z * wb.z + cv.w * wb.w;
  }
  s = (s > 0.f) ? s : 0.2f * s;  // leaky_relu(0.2)
  sbuf[i] = s;
  atomicMax(&mmax[b * N_ + node], f2mono(s));
}

__global__ void k_att_norm(const int* __restrict__ src,
                           const int* __restrict__ dst,
                           const float* __restrict__ sbuf,
                           const unsigned* __restrict__ mmax,
                           float* __restrict__ exbuf,
                           float* __restrict__ den) {
  long i = (long)blockIdx.x * blockDim.x + threadIdx.x;
  const long P = (long)B_ * 2 * E_;
  if (i >= P) return;
  int b = (int)(i / (2 * E_));
  int p = (int)(i % (2 * E_));
  int node = (p < E_) ? src[p] : dst[p - E_];
  float mval = mono2f(mmax[b * N_ + node]);
  float ex = expf(sbuf[i] - mval);
  exbuf[i] = ex;
  atomicAdd(&den[b * N_ + node], ex);
}

__global__ void k_att_agg(const float* __restrict__ edge_emb,
                          const int* __restrict__ src,
                          const int* __restrict__ dst,
                          const float* __restrict__ exbuf,
                          const float* __restrict__ den,
                          float* __restrict__ agg) {
  long i = (long)blockIdx.x * blockDim.x + threadIdx.x;
  const long T = (long)B_ * 2 * E_ * D_;
  if (i >= T) return;
  int d = (int)(i & (D_ - 1));
  long pp = i >> 7;
  int b = (int)(pp / (2 * E_));
  int p = (int)(pp % (2 * E_));
  int node = (p < E_) ? src[p] : dst[p - E_];
  int e = (p < E_) ? p : p - E_;
  float w = exbuf[pp] / den[b * N_ + node];
  atomicAdd(&agg[((long)b * N_ + node) * D_ + d],
            w * edge_emb[((long)b * E_ + e) * D_ + d]);
}

__global__ void k_node_update(const float* __restrict__ agg,
                              const float* __restrict__ deg,
                              float* __restrict__ node_emb) {
  long i = (long)blockIdx.x * blockDim.x + threadIdx.x;
  const long T = (long)B_ * N_ * D_;
  if (i >= T) return;
  int n = (int)((i >> 7) % N_);
  if (deg[n] > 0.f) node_emb[i] = gelu_f(agg[i]);
}

// -------------------------------- decoder ----------------------------------
// Block = 64 threads = 2 waves; wave handles a 16-edge tile (256 WMMAs).

__global__ __launch_bounds__(64) void k_decoder(
    const float* __restrict__ node_emb, const float* __restrict__ edge_emb,
    const int* __restrict__ src, const int* __restrict__ dst,
    const _Float16* __restrict__ dw1t, const float* __restrict__ db1,
    const _Float16* __restrict__ dw2t, const float* __restrict__ db2,
    const float* __restrict__ dw3, const float* __restrict__ db3,
    float* __restrict__ out) {
  __shared__ __attribute__((aligned(16))) _Float16 ldsA[2][16][384];
  __shared__ __attribute__((aligned(16))) _Float16 ldsZ1[2][16][256];
  __shared__ __attribute__((aligned(16))) _Float16 ldsZ2[2][16][128];

  int wave = threadIdx.x >> 5;
  int lane = threadIdx.x & 31;
  int half = lane >> 4;
  int m = lane & 15;
  long row0 = ((long)blockIdx.x * 2 + wave) * 16;

  // Stage A = [h_base | h_quote | edge_emb] (16x384) as f16 in LDS.
  for (int idx = lane; idx < 16 * 384; idx += 32) {
    int r = idx / 384, k = idx % 384;
    long g = row0 + r;
    int b = (int)(g / E_);
    int e = (int)(g % E_);
    float v;
    if (k < 128)
      v = node_emb[((long)b * N_ + src[e]) * D_ + k];
    else if (k < 256)
      v = node_emb[((long)b * N_ + dst[e]) * D_ + (k - 128)];
    else
      v = edge_emb[g * D_ + (k - 256)];
    ldsA[wave][r][k] = (_Float16)v;
  }
  __builtin_prefetch(edge_emb + (row0 + 16) * D_, 0, 1);
  __syncthreads();

  // ---- layer1: (16x384) @ (384x256), GELU ----
  v16h a1[12];
#pragma unroll
  for (int q = 0; q < 12; ++q)
    a1[q] = frag16(&ldsA[wave][m][0], 32 * q, half);

#pragma unroll 1
  for (int t = 0; t < 16; ++t) {
    const _Float16* wrow = dw1t + (long)(t * 16 + m) * 384;
    v8f acc = {};
#pragma unroll
    for (int q = 0; q < 12; ++q)
      acc = wmma_f16(a1[q], frag16(wrow, 32 * q, half), acc);
    float bb = db1[t * 16 + m];
#pragma unroll
    for (int e = 0; e < 8; ++e)
      ldsZ1[wave][half * 8 + e][t * 16 + m] = (_Float16)gelu_f(acc[e] + bb);
  }
  __syncthreads();

  // ---- layer2: (16x256) @ (256x128), GELU ----
  v16h a2[8];
#pragma unroll
  for (int q = 0; q < 8; ++q)
    a2[q] = frag16(&ldsZ1[wave][m][0], 32 * q, half);

#pragma unroll 1
  for (int t = 0; t < 8; ++t) {
    const _Float16* wrow = dw2t + (long)(t * 16 + m) * 256;
    v8f acc = {};
#pragma unroll
    for (int q = 0; q < 8; ++q)
      acc = wmma_f16(a2[q], frag16(wrow, 32 * q, half), acc);
    float bb = db2[t * 16 + m];
#pragma unroll
    for (int e = 0; e < 8; ++e)
      ldsZ2[wave][half * 8 + e][t * 16 + m] = (_Float16)gelu_f(acc[e] + bb);
  }
  __syncthreads();

  // ---- layer3: (16x128) @ (128x1) GEMV, two half-lanes per row ----
  float part = 0.f;
  const v8h* zp = (const v8h*)&ldsZ2[wave][m][half * 64];
#pragma unroll
  for (int cch = 0; cch < 8; ++cch) {
    v8h z = zp[cch];
#pragma unroll
    for (int j = 0; j < 8; ++j)
      part += (float)z[j] * dw3[half * 64 + cch * 8 + j];
  }
  part += __shfl_xor(part, 16, 32);
  if (half == 0) out[row0 + m] = part + db3[0];
}

// ------------------------------ host launcher ------------------------------

extern "C" void kernel_launch(void* const* d_in, const int* in_sizes, int n_in,
                              void* d_out, int out_size, void* d_ws,
                              size_t ws_size, hipStream_t stream) {
  (void)in_sizes; (void)n_in; (void)out_size; (void)ws_size;

  const float* edge_features = (const float*)d_in[0];
  const float* enc_w1 = (const float*)d_in[1];
  const float* enc_b1 = (const float*)d_in[2];
  const float* enc_g1 = (const float*)d_in[3];
  const float* enc_be1 = (const float*)d_in[4];
  const float* enc_w2 = (const float*)d_in[5];
  const float* enc_b2 = (const float*)d_in[6];
  const float* enc_g2 = (const float*)d_in[7];
  const float* enc_be2 = (const float*)d_in[8];
  const float* watt1 = (const float*)d_in[9];
  const float* batt1 = (const float*)d_in[10];
  const float* watt2 = (const float*)d_in[11];
  const float* batt2 = (const float*)d_in[12];
  const float* dec_w1 = (const float*)d_in[13];
  const float* dec_b1 = (const float*)d_in[14];
  const float* dec_w2 = (const float*)d_in[15];
  const float* dec_b2 = (const float*)d_in[16];
  const float* dec_w3 = (const float*)d_in[17];
  const float* dec_b3 = (const float*)d_in[18];
  const int* src = (const int*)d_in[19];
  const int* dst = (const int*)d_in[20];
  float* out = (float*)d_out;

  // Carve workspace.
  char* ws = (char*)d_ws;
  size_t off = 0;
  auto carve = [&](size_t bytes) -> void* {
    void* p = ws + off;
    off += (bytes + 255) & ~(size_t)255;
    return p;
  };
  float* edge_emb = (float*)carve((size_t)B_ * E_ * D_ * 4);
  float* node_emb = (float*)carve((size_t)B_ * N_ * D_ * 4);
  float* agg = (float*)carve((size_t)B_ * N_ * D_ * 4);
  unsigned* mmax = (unsigned*)carve((size_t)B_ * N_ * 4);
  float* den = (float*)carve((size_t)B_ * N_ * 4);
  float* deg = (float*)carve((size_t)N_ * 4);
  float* sbuf = (float*)carve((size_t)B_ * 2 * E_ * 4);
  float* exbuf = (float*)carve((size_t)B_ * 2 * E_ * 4);
  _Float16* w1t = (_Float16*)carve((size_t)128 * 32 * 2);     // [N=128][Kpad=32]
  _Float16* w2t = (_Float16*)carve((size_t)128 * 128 * 2);    // [128][128]
  _Float16* dw1t = (_Float16*)carve((size_t)256 * 384 * 2);   // [256][384]
  _Float16* dw2t = (_Float16*)carve((size_t)128 * 256 * 2);   // [128][256]

  auto cdiv = [](long a, long b) { return (int)((a + b - 1) / b); };

  // Repack weights: (K x N) f32 -> transposed f16 [N][Kpad], zero-padded K.
  k_repack_wT<<<cdiv(128 * 32, 256), 256, 0, stream>>>(enc_w1, w1t, 24, 128, 32);
  k_repack_wT<<<cdiv(128 * 128, 256), 256, 0, stream>>>(enc_w2, w2t, 128, 128, 128);
  k_repack_wT<<<cdiv(256 * 384, 256), 256, 0, stream>>>(dec_w1, dw1t, 384, 256, 384);
  k_repack_wT<<<cdiv(128 * 256, 256), 256, 0, stream>>>(dec_w2, dw2t, 256, 128, 256);

  // Degrees (shared across batches).
  k_fill_u32<<<cdiv(N_, 256), 256, 0, stream>>>((unsigned*)deg, 0u, N_);
  k_deg<<<cdiv(2 * E_, 256), 256, 0, stream>>>(src, dst, deg);

  // Edge encoder (WMMA).
  k_encoder<<<(B_ * E_) / (16 * 4), 128, 0, stream>>>(
      edge_features, w1t, enc_b1, enc_g1, enc_be1, w2t, enc_b2, enc_g2,
      enc_be2, edge_emb);

  // node_emb starts at zero.
  k_fill_u32<<<cdiv((long)B_ * N_ * D_, 256), 256, 0, stream>>>(
      (unsigned*)node_emb, 0u, B_ * N_ * D_);

  const long P = (long)B_ * 2 * E_;
  for (int r = 0; r < 2; ++r) {
    const float* watt = r ? watt2 : watt1;
    const float* batt = r ? batt2 : batt1;
    k_fill_u32<<<cdiv(B_ * N_, 256), 256, 0, stream>>>(mmax, 0u, B_ * N_);
    k_fill_u32<<<cdiv(B_ * N_, 256), 256, 0, stream>>>((unsigned*)den, 0u,
                                                       B_ * N_);
    k_fill_u32<<<cdiv((long)B_ * N_ * D_, 256), 256, 0, stream>>>(
        (unsigned*)agg, 0u, B_ * N_ * D_);
    k_att_score<<<cdiv(P, 256), 256, 0, stream>>>(node_emb, edge_emb, src, dst,
                                                  watt, batt, sbuf, mmax);
    k_att_norm<<<cdiv(P, 256), 256, 0, stream>>>(src, dst, sbuf, mmax, exbuf,
                                                 den);
    k_att_agg<<<cdiv(P * D_, 256), 256, 0, stream>>>(edge_emb, src, dst, exbuf,
                                                     den, agg);
    k_node_update<<<cdiv((long)B_ * N_ * D_, 256), 256, 0, stream>>>(agg, deg,
                                                                     node_emb);
  }

  // Decoder (WMMA) -> vel.
  k_decoder<<<(B_ * E_) / (16 * 2), 64, 0, stream>>>(
      node_emb, edge_emb, src, dst, dw1t, dec_b1, dw2t, dec_b2, dec_w3, dec_b3,
      out);
}
